// AirMPNN_58531814310350
// MI455X (gfx1250) — matrix operand order repaired and checked by reading, not compiled, not run on hardware
//
#include <hip/hip_runtime.h>
#include <hip/hip_bf16.h>

typedef __attribute__((ext_vector_type(16))) _Float16 v16h;
typedef __attribute__((ext_vector_type(8)))  _Float16 v8h;
typedef __attribute__((ext_vector_type(8)))  float    v8f;

#define XPAD 16   // padded node-feature row stride (floats)

__device__ __forceinline__ float sigmoidf_(float z) {
    return 1.0f / (1.0f + __expf(-z));
}
__device__ __forceinline__ float reluf_(float x) {
    return fmaxf(x, 0.0f);
}

// ---- one-time weight prep: f32 -> f16, K padded to 32; b1 folded into row 9 ----
__global__ void prep_weights(const float* __restrict__ W1,
                             const float* __restrict__ b1,
                             const float* __restrict__ W2,
                             _Float16* __restrict__ W1h,   // [32*32] row k, col j
                             _Float16* __restrict__ W2h)   // [32*32]
{
    int t = threadIdx.x;            // 0..1023
    int k = t >> 5, j = t & 31;
    float w1;
    if (k < 9)       w1 = W1[k * 32 + j];
    else if (k == 9) w1 = b1[j];          // bias row: pairs with B1[K=9] = 1.0
    else             w1 = 0.f;
    W1h[t] = (_Float16)w1;
    W2h[t] = (_Float16)W2[t];
}

__global__ void pad_x(const float* __restrict__ x, float* __restrict__ xp, int n) {
    int i = blockIdx.x * blockDim.x + threadIdx.x;
    if (i >= n * XPAD) return;
    int node = i / XPAD, k = i % XPAD;
    xp[i] = (k < 9) ? x[node * 9 + k] : 0.f;
}

__global__ void zero_f32(float* __restrict__ p, int n) {
    int i = blockIdx.x * blockDim.x + threadIdx.x;
    if (i < n) p[i] = 0.f;
}

// ---- edge MLP + scatter-add (transposed WMMA, feature-permuted hidden layer) ----
// h1^T = W1b^T (A) x [x;1]^T (B)  -> f16 D via V_WMMA_F16_16X16X32_F16
// h2^T = W2p^T (A) x h1^T (B)     -> f32 D, bias as C-init
// msg  = sigmoid(<relu(h2), W3> + b3) * attr  -> atomic scatter-add
// D->B handoff is lane-local identity: B2[j] = relu(D_{j>>3}[j&7]); the implied
// feature permutation is folded into the loop-invariant aW2 preload.
__global__ void __launch_bounds__(256)
edge_kernel(const float* __restrict__ xp,        // [N,16] padded f32
            const int*   __restrict__ srcIdx,    // [E]
            const int*   __restrict__ dstIdx,    // [E]
            const float* __restrict__ attr,      // [E]
            const _Float16* __restrict__ W1h,    // [32,32] f16 (K-padded, b1 in row 9)
            const _Float16* __restrict__ W2h,    // [32,32] f16
            const float* __restrict__ W3,        // [32] f32
            const float* __restrict__ b2,
            const float* __restrict__ b3,
            float* __restrict__ agg,             // [N]
            int E)
{
    const int  lane = threadIdx.x & 31;
    const int  wave = threadIdx.x >> 5;
    const int  col  = lane & 15;                 // edge-in-tile (B/D column)
    const bool lo   = lane < 16;
    const int  rb   = lo ? 0 : 8;                // D-matrix element row base

    // --- A operands: weight tiles in the 16-bit A layout (loop-invariant) ---
    //     A element j on this lane sits at hardware-K:  kk = ((j<8)?j:j+8) + (lo?0:8)
    v16h aW1[2], aW2[2];
    #pragma unroll
    for (int t = 0; t < 2; ++t) {
        const int feat = col + 16 * t;           // M row = output feature
        #pragma unroll
        for (int j = 0; j < 16; ++j) {
            const int kk = ((j < 8) ? j : j + 8) + (lo ? 0 : 8);
            aW1[t][j] = W1h[(j + (lo ? 0 : 16)) * 32 + feat];     // W1^T[feat][k] (+bias row 9)
            // hidden feature living at hardware-K slot kk is
            //   r = kk
            const int r = kk;
            aW2[t][j] = W2h[r * 32 + feat];      // W2^T[feat][r]
        }
    }
    // --- layer-2 bias as WMMA C-init: c[i] corresponds to feature (rb+i+16t) ---
    v8f cb2[2];
    #pragma unroll
    for (int t = 0; t < 2; ++t)
        #pragma unroll
        for (int i = 0; i < 8; ++i)
            cb2[t][i] = b2[rb + i + 16 * t];
    // --- W3 slices covered by this lane's D elements ---
    float w3a[8], w3b[8];
    #pragma unroll
    for (int i = 0; i < 8; ++i) { w3a[i] = W3[rb + i]; w3b[i] = W3[rb + i + 16]; }
    const float b3s = b3[0];

    const int nTiles = (E + 15) >> 4;
    const int waveId = blockIdx.x * 8 + wave;
    const int nWaves = gridDim.x * 8;

    for (int tile = waveId; tile < nTiles; tile += nWaves) {
        const int e0 = tile << 4;
        const int e  = e0 + col;
        const int ee = (e < E) ? e : (E - 1);

        // --- B1 = [x;1]^T column for this lane's edge; hi-lane rows (K16..31) zero ---
        v16h B1;
        #pragma unroll
        for (int j = 0; j < 16; ++j) B1[j] = (_Float16)0.f;
        if (lo) {
            const int s = srcIdx[ee];
            const float4* px = (const float4*)(xp + (size_t)s * XPAD);
            float4 f0 = px[0], f1 = px[1], f2 = px[2];   // feats 0..11 (9..15 are 0 pad)
            B1[0]=(_Float16)f0.x; B1[1]=(_Float16)f0.y; B1[2]=(_Float16)f0.z; B1[3]=(_Float16)f0.w;
            B1[4]=(_Float16)f1.x; B1[5]=(_Float16)f1.y; B1[6]=(_Float16)f1.z; B1[7]=(_Float16)f1.w;
            B1[8]=(_Float16)f2.x;                        // feature 8
            B1[9]=(_Float16)1.0f;                        // bias input (row 9 of W1h)
        }

        // --- layer 1: f16 D = W1b^T x [x;1]^T  (C = inline 0) ---
        v8h hz0 = {}, hz1 = {};
        v8h D0h = __builtin_amdgcn_wmma_f16_16x16x32_f16(false, aW1[0], false, B1, (short)0, hz0, false, false);
        v8h D1h = __builtin_amdgcn_wmma_f16_16x16x32_f16(false, aW1[1], false, B1, (short)0, hz1, false, false);

        // --- ReLU on f16 vectors (lowers to v_pk_max_f16), identity handoff ---
        v8h zero8 = {};
        v8h r0 = __builtin_elementwise_max(D0h, zero8);
        v8h r1 = __builtin_elementwise_max(D1h, zero8);
        v16h B2;
        #pragma unroll
        for (int i = 0; i < 8; ++i) {
            B2[i]     = r0[i];
            B2[8 + i] = r1[i];
        }

        // --- layer 2: f32 D = W2p^T x h1^T + b2 ---
        v8f E0 = cb2[0], E1 = cb2[1];
        E0 = __builtin_amdgcn_wmma_f32_16x16x32_f16(false, aW2[0], false, B2, (short)0, E0, false, false);
        E1 = __builtin_amdgcn_wmma_f32_16x16x32_f16(false, aW2[1], false, B2, (short)0, E1, false, false);

        // --- layer 3: per-edge dot(relu(h2), W3); each lane covers 16 feats ---
        float partial = 0.f;
        #pragma unroll
        for (int i = 0; i < 8; ++i) {
            partial += reluf_(E0[i]) * w3a[i];
            partial += reluf_(E1[i]) * w3b[i];
        }
        const int pbits = __builtin_amdgcn_ds_swizzle(__float_as_int(partial), 0x401f); // SWAPX16
        const float z = partial + __int_as_float(pbits) + b3s;

        if (lo && e < E) {
            const float m = sigmoidf_(z) * attr[e];
            atomicAdd(agg + dstIdx[e], m);
        }
    }
}

// ---- per-node combine MLP: t=[x(9),agg] -> 16 -> 8 ----
__global__ void node_kernel(const float* __restrict__ xp,
                            const float* __restrict__ agg,
                            const float* __restrict__ V1, const float* __restrict__ c1,
                            const float* __restrict__ V2, const float* __restrict__ c2,
                            float* __restrict__ xout, int n)
{
    int i = blockIdx.x * blockDim.x + threadIdx.x;
    if (i >= n) return;
    float t[10];
    #pragma unroll
    for (int k = 0; k < 9; ++k) t[k] = xp[(size_t)i * XPAD + k];
    t[9] = agg[i];                         // (agg - 0) / 1
    float h[16];
    #pragma unroll
    for (int j = 0; j < 16; ++j) {
        float acc = c1[j];
        #pragma unroll
        for (int k = 0; k < 10; ++k) acc += t[k] * V1[k * 16 + j];
        h[j] = reluf_(acc);
    }
    float* out = xout + (size_t)i * XPAD;
    out[0] = t[0];
    #pragma unroll
    for (int q = 0; q < 8; ++q) {
        float acc = c2[q];
        #pragma unroll
        for (int j = 0; j < 16; ++j) acc += h[j] * V2[j * 8 + q];
        out[1 + q] = reluf_(acc);
    }
    #pragma unroll
    for (int k = 9; k < XPAD; ++k) out[k] = 0.f;
}

// ---- readout head: 8 -> 16 -> 1, sigmoid ----
__global__ void head_kernel(const float* __restrict__ xp,
                            const float* __restrict__ H1, const float* __restrict__ d1,
                            const float* __restrict__ H2, const float* __restrict__ d2,
                            float* __restrict__ out, int n)
{
    int i = blockIdx.x * blockDim.x + threadIdx.x;
    if (i >= n) return;
    float f[8];
    #pragma unroll
    for (int k = 0; k < 8; ++k) f[k] = xp[(size_t)i * XPAD + 1 + k];
    float z = d2[0];
    #pragma unroll
    for (int j = 0; j < 16; ++j) {
        float acc = d1[j];
        #pragma unroll
        for (int k = 0; k < 8; ++k) acc += f[k] * H1[k * 16 + j];
        z += reluf_(acc) * H2[j];
    }
    out[i] = sigmoidf_(z);
}

extern "C" void kernel_launch(void* const* d_in, const int* in_sizes, int n_in,
                              void* d_out, int out_size, void* d_ws, size_t ws_size,
                              hipStream_t stream)
{
    const float* x    = (const float*)d_in[0];
    const float* attr = (const float*)d_in[1];
    const int*   ei   = (const int*)d_in[2];
    const float *W1 = (const float*)d_in[3],  *b1 = (const float*)d_in[4];
    const float *W2 = (const float*)d_in[5],  *b2 = (const float*)d_in[6];
    const float *W3 = (const float*)d_in[7],  *b3 = (const float*)d_in[8];
    const float *V1 = (const float*)d_in[9],  *c1 = (const float*)d_in[10];
    const float *V2 = (const float*)d_in[11], *c2 = (const float*)d_in[12];
    const float *H1 = (const float*)d_in[13], *d1 = (const float*)d_in[14];
    const float *H2 = (const float*)d_in[15], *d2 = (const float*)d_in[16];

    const int N = in_sizes[0] / 9;
    const int E = in_sizes[1];
    const int* srcIdx = ei;        // edge_index[0]
    const int* dstIdx = ei + E;    // edge_index[1]

    char* ws = (char*)d_ws;
    size_t off = 0;
    auto alloc = [&](size_t bytes) -> void* {
        void* p = ws + off;
        off = (off + bytes + 255) & ~(size_t)255;
        return p;
    };
    float*    xpA = (float*)alloc((size_t)N * XPAD * sizeof(float));
    float*    xpB = (float*)alloc((size_t)N * XPAD * sizeof(float));
    float*    agg = (float*)alloc((size_t)N * sizeof(float));
    _Float16* W1h = (_Float16*)alloc(1024 * sizeof(_Float16));
    _Float16* W2h = (_Float16*)alloc(1024 * sizeof(_Float16));
    (void)ws_size;

    prep_weights<<<1, 1024, 0, stream>>>(W1, b1, W2, W1h, W2h);
    {
        int tot = N * XPAD;
        pad_x<<<(tot + 255) / 256, 256, 0, stream>>>(x, xpA, N);
    }

    float* cur = xpA;
    float* nxt = xpB;
    for (int r = 0; r < 3; ++r) {
        zero_f32<<<(N + 255) / 256, 256, 0, stream>>>(agg, N);
        edge_kernel<<<1024, 256, 0, stream>>>(cur, srcIdx, dstIdx, attr,
                                              W1h, W2h, W3, b2, b3, agg, E);
        node_kernel<<<(N + 255) / 256, 256, 0, stream>>>(cur, agg, V1, c1, V2, c2, nxt, N);
        float* tmp = cur; cur = nxt; nxt = tmp;
    }
    head_kernel<<<(N + 255) / 256, 256, 0, stream>>>(cur, H1, d1, H2, d2, (float*)d_out, N);
}
